// SigLIP2MultiPositiveBCELoss_36799279792929
// MI455X (gfx1250) — matrix operand order repaired and checked by reading, not compiled, not run on hardware
//
#include <hip/hip_runtime.h>
#include <hip/hip_bf16.h>
#include <stdint.h>

// MI455X / gfx1250: wave32, WMMA 16x16x32 bf16, async global->LDS pipeline.
// Split-bf16 (hi+lo) emulated-f32 GEMM fused with SigLIP BCE loss.

typedef __attribute__((ext_vector_type(16))) __bf16 v16bf;
typedef __attribute__((ext_vector_type(8)))  float  v8f;
typedef __attribute__((ext_vector_type(4)))  int    v4i;

#define D_DIM   512
#define TILE    128          // workgroup tile (M and N)
#define KSTEP   32           // k per pipeline stage (one WMMA k)
#define NSTEP   (D_DIM / KSTEP)
#define RS      40           // LDS row stride in bf16 elems (80B: 16B-aligned, 20-dword bank step)
#define MAT_ELEMS (TILE * RS)      // 5120 elems per matrix tile
#define BUF_ELEMS (4 * MAT_ELEMS)  // Ah | Al | Bh | Bl
#define LDS_BYTES (2 * BUF_ELEMS * 2)  // double buffered, 81920 B (dynamic LDS)

#if defined(__HIP_DEVICE_COMPILE__) && \
    __has_builtin(__builtin_amdgcn_global_load_async_to_lds_b128) && \
    __has_builtin(__builtin_amdgcn_s_wait_asynccnt)
#define USE_ASYNC 1
// Pointer types matching the builtin signature: (v4i AS1*, v4i AS3*, imm, imm)
typedef __attribute__((address_space(1))) v4i* as1_v4i_ptr;
typedef __attribute__((address_space(3))) v4i* as3_v4i_ptr;
#else
#define USE_ASYNC 0
#endif

__device__ __forceinline__ unsigned short f32_to_bf16_rne(float x) {
  unsigned int u = __float_as_uint(x);
  u += 0x7FFFu + ((u >> 16) & 1u);   // round-to-nearest-even
  return (unsigned short)(u >> 16);
}
__device__ __forceinline__ float bf16_to_f32(unsigned short h) {
  return __uint_as_float(((unsigned int)h) << 16);
}

union FragU { uint4 q[2]; v16bf v; };

// -------- Kernel 1: L2-normalize rows, emit bf16 hi/lo split --------
__global__ __launch_bounds__(256) void norm_split_kernel(
    const float* __restrict__ V, const float* __restrict__ T,
    unsigned short* __restrict__ vh, unsigned short* __restrict__ vl,
    unsigned short* __restrict__ th, unsigned short* __restrict__ tl,
    int Bv) {
  int row = blockIdx.x;
  const float* src;
  unsigned short *dh, *dl;
  if (row < Bv) {
    src = V + (size_t)row * D_DIM;
    dh = vh + (size_t)row * D_DIM; dl = vl + (size_t)row * D_DIM;
  } else {
    int r = row - Bv;
    src = T + (size_t)r * D_DIM;
    dh = th + (size_t)r * D_DIM; dl = tl + (size_t)r * D_DIM;
  }
  int t0 = threadIdx.x, t1 = threadIdx.x + 256;   // 512 elems / 256 threads
  float x0 = src[t0], x1 = src[t1];
  float ss = x0 * x0 + x1 * x1;
  for (int o = 16; o > 0; o >>= 1) ss += __shfl_xor(ss, o, 32);
  __shared__ float red[8];
  int lane = threadIdx.x & 31, wv = threadIdx.x >> 5;
  if (lane == 0) red[wv] = ss;
  __syncthreads();
  if (wv == 0) {
    float s = (lane < 8) ? red[lane] : 0.0f;
    for (int o = 4; o > 0; o >>= 1) s += __shfl_xor(s, o, 32);
    if (lane == 0) red[0] = s;
  }
  __syncthreads();
  float inv = 1.0f / fmaxf(sqrtf(red[0]), 1e-12f);   // matches F.normalize eps
  float y0 = x0 * inv;
  unsigned short h0 = f32_to_bf16_rne(y0);
  dh[t0] = h0; dl[t0] = f32_to_bf16_rne(y0 - bf16_to_f32(h0));
  float y1 = x1 * inv;
  unsigned short h1 = f32_to_bf16_rne(y1);
  dh[t1] = h1; dl[t1] = f32_to_bf16_rne(y1 - bf16_to_f32(h1));
}

// Coalesced global -> LDS tile copy for one k-stage.
// 4 matrices x 128 rows x 4 chunks(16B) = 2048 chunks; 256 threads x 8 chunks.
// Consecutive lanes take consecutive chunks -> 64B-contiguous per row, 8 rows/wave.
__device__ __forceinline__ void issue_tile_copies(
    const unsigned short* __restrict__ vh, const unsigned short* __restrict__ vl,
    const unsigned short* __restrict__ th, const unsigned short* __restrict__ tl,
    unsigned short* smem, int buf, int kb, int rowG0, int colG0, int tid) {
  const unsigned short* bases[4] = {
      vh + (size_t)rowG0 * D_DIM, vl + (size_t)rowG0 * D_DIM,
      th + (size_t)colG0 * D_DIM, tl + (size_t)colG0 * D_DIM };
#pragma unroll
  for (int j = 0; j < 8; ++j) {
    int c   = j * 256 + tid;
    int kc  = c & 3;              // 16B chunk within the 64B k-window
    int row = (c >> 2) & 127;     // row within tile
    int m   = c >> 9;             // matrix: 0=Ah 1=Al 2=Bh 3=Bl
    const unsigned short* gp = bases[m] + (size_t)row * D_DIM + kb + kc * 8;
    unsigned short* lp = smem + (size_t)buf * BUF_ELEMS + m * MAT_ELEMS + row * RS + kc * 8;
#if USE_ASYNC
    __builtin_amdgcn_global_load_async_to_lds_b128(
        (as1_v4i_ptr)(gp), (as3_v4i_ptr)(lp), 0, 0);
#else
    *(uint4*)lp = *(const uint4*)gp;
#endif
  }
}

// A fragment (16x32 bf16): lane m=lane&15 holds row m; chunks at k=8*sel and 16+8*sel.
__device__ __forceinline__ v16bf lds_a_frag(const unsigned short* base, int row, int halfsel) {
  FragU u;
  const unsigned short* p = base + row * RS + halfsel * 8;
  u.q[0] = *(const uint4*)(p);
  u.q[1] = *(const uint4*)(p + 16);
  return u.v;
}
// B fragment (32x16 bf16): lane n=lane&15 holds column n; chunks at k=8*sel and 16+8*sel.
__device__ __forceinline__ v16bf lds_b_frag(const unsigned short* base, int row, int halfsel) {
  FragU u;
  const unsigned short* p = base + row * RS + halfsel * 8;
  u.q[0] = *(const uint4*)(p);
  u.q[1] = *(const uint4*)(p + 16);
  return u.v;
}

// -------- Kernel 2: async-LDS double-buffered split-bf16 WMMA GEMM + fused BCE --------
__global__ __launch_bounds__(256) void siglip_wmma_loss_kernel(
    const unsigned short* __restrict__ vh, const unsigned short* __restrict__ vl,
    const unsigned short* __restrict__ th, const unsigned short* __restrict__ tl,
    const float* __restrict__ log_temp, const float* __restrict__ bias,
    float* __restrict__ partials) {
  extern __shared__ __align__(16) unsigned short smem[];

  const int tid     = threadIdx.x;
  const int lane    = tid & 31;
  const int wv      = tid >> 5;          // 8 waves: 2 (M) x 4 (N)
  const int halfsel = lane >> 4;
  const int l15     = lane & 15;

  const int rowG0 = blockIdx.y * TILE;
  const int colG0 = blockIdx.x * TILE;
  const int rBase = rowG0 + (wv >> 2) * 64;   // wave owns 64x32
  const int cBase = colG0 + (wv & 3) * 32;
  const int rLoc  = (wv >> 2) * 64;           // wave base within LDS tile
  const int cLoc  = (wv & 3) * 32;

  v8f acc[4][2];
  const v8f vz = {0.f, 0.f, 0.f, 0.f, 0.f, 0.f, 0.f, 0.f};
#pragma unroll
  for (int mi = 0; mi < 4; ++mi)
#pragma unroll
    for (int ni = 0; ni < 2; ++ni) acc[mi][ni] = vz;

#if USE_ASYNC
  issue_tile_copies(vh, vl, th, tl, smem, 0, 0, rowG0, colG0, tid);  // prologue
#endif

  for (int step = 0; step < NSTEP; ++step) {
    const int buf = step & 1;
    const int kb  = step * KSTEP;
#if USE_ASYNC
    __builtin_amdgcn_s_wait_asynccnt(0);   // my copies for this stage have landed
    __syncthreads();                       // everyone's landed; prior compute done
    if (step + 1 < NSTEP)                  // prefetch next stage into other buffer
      issue_tile_copies(vh, vl, th, tl, smem, buf ^ 1, kb + KSTEP, rowG0, colG0, tid);
#else
    issue_tile_copies(vh, vl, th, tl, smem, buf, kb, rowG0, colG0, tid);
    __syncthreads();
#endif

    const unsigned short* Ah = smem + (size_t)buf * BUF_ELEMS + 0 * MAT_ELEMS;
    const unsigned short* Al = smem + (size_t)buf * BUF_ELEMS + 1 * MAT_ELEMS;
    const unsigned short* Bh = smem + (size_t)buf * BUF_ELEMS + 2 * MAT_ELEMS;
    const unsigned short* Bl = smem + (size_t)buf * BUF_ELEMS + 3 * MAT_ELEMS;

    v16bf aH[4], aL[4], bH[2], bL[2];
#pragma unroll
    for (int mi = 0; mi < 4; ++mi) {
      int r = rLoc + mi * 16 + l15;
      aH[mi] = lds_a_frag(Ah, r, halfsel);
      aL[mi] = lds_a_frag(Al, r, halfsel);
    }
#pragma unroll
    for (int ni = 0; ni < 2; ++ni) {
      int c = cLoc + ni * 16 + l15;
      bH[ni] = lds_b_frag(Bh, c, halfsel);
      bL[ni] = lds_b_frag(Bl, c, halfsel);
    }
#pragma unroll
    for (int mi = 0; mi < 4; ++mi)
#pragma unroll
      for (int ni = 0; ni < 2; ++ni) {
        // sim ~= hi*lo + lo*hi + hi*hi  (lo*lo dropped, ~2^-16 relative)
        acc[mi][ni] = __builtin_amdgcn_wmma_f32_16x16x32_bf16(
            false, aH[mi], false, bL[ni], (short)0, acc[mi][ni], false, false);
        acc[mi][ni] = __builtin_amdgcn_wmma_f32_16x16x32_bf16(
            false, aL[mi], false, bH[ni], (short)0, acc[mi][ni], false, false);
        acc[mi][ni] = __builtin_amdgcn_wmma_f32_16x16x32_bf16(
            false, aH[mi], false, bH[ni], (short)0, acc[mi][ni], false, false);
      }
  }

  // Fused epilogue: logits -> clamp -> stable BCE-with-logits -> local sum
  const float invT = __expf(-log_temp[0]);   // 1 / exp(log_temp)
  const float bs   = bias[0];
  float lsum = 0.0f;
#pragma unroll
  for (int mi = 0; mi < 4; ++mi)
#pragma unroll
    for (int ni = 0; ni < 2; ++ni) {
      int c = cBase + ni * 16 + l15;
#pragma unroll
      for (int j = 0; j < 8; ++j) {
        // C/D layout: VGPR j -> M = j (lanes 0-15) or 8+j (lanes 16-31), N = lane&15
        int r = rBase + mi * 16 + 8 * halfsel + j;
        float z = acc[mi][ni][j] * invT + bs;
        z = fminf(fmaxf(z, -30.0f), 30.0f);
        float bce = fmaxf(z, 0.0f) + log1pf(__expf(-fabsf(z)));
        if (r == c) bce -= z;   // diagonal positive label
        lsum += bce;
      }
    }

  for (int o = 16; o > 0; o >>= 1) lsum += __shfl_xor(lsum, o, 32);
  __shared__ float red[8];
  if (lane == 0) red[wv] = lsum;
  __syncthreads();
  if (tid == 0) {
    float t = 0.0f;
    for (int i = 0; i < 8; ++i) t += red[i];
    partials[blockIdx.y * gridDim.x + blockIdx.x] = t;   // no atomics: deterministic
  }
}

// -------- Kernel 3: deterministic final reduction --------
__global__ __launch_bounds__(256) void finalize_kernel(
    const float* __restrict__ partials, int n, float scale, float* __restrict__ out) {
  float s = 0.0f;
  for (int i = threadIdx.x; i < n; i += 256) s += partials[i];
  for (int o = 16; o > 0; o >>= 1) s += __shfl_xor(s, o, 32);
  __shared__ float red[8];
  int lane = threadIdx.x & 31, wv = threadIdx.x >> 5;
  if (lane == 0) red[wv] = s;
  __syncthreads();
  if (threadIdx.x == 0) {
    float t = 0.0f;
    for (int i = 0; i < 8; ++i) t += red[i];
    out[0] = t * scale;
  }
}

extern "C" void kernel_launch(void* const* d_in, const int* in_sizes, int n_in,
                              void* d_out, int out_size, void* d_ws, size_t ws_size,
                              hipStream_t stream) {
  const float* V        = (const float*)d_in[0];
  const float* T        = (const float*)d_in[1];
  const float* log_temp = (const float*)d_in[2];
  const float* bias     = (const float*)d_in[3];
  float* out            = (float*)d_out;

  const int Bv = in_sizes[0] / D_DIM;   // 8192
  const int Bt = in_sizes[1] / D_DIM;   // 8192

  // Workspace layout: v_hi | v_lo | t_hi | t_lo (bf16) | tile partials (f32)
  size_t nV = (size_t)Bv * D_DIM, nT = (size_t)Bt * D_DIM;
  unsigned short* vh = (unsigned short*)d_ws;
  unsigned short* vl = vh + nV;
  unsigned short* th = vl + nV;
  unsigned short* tl = th + nT;
  float* partials    = (float*)(tl + nT);

  const int gM = Bv / TILE, gN = Bt / TILE;

  hipLaunchKernelGGL(norm_split_kernel, dim3(Bv + Bt), dim3(256), 0, stream,
                     V, T, vh, vl, th, tl, Bv);
  hipLaunchKernelGGL(siglip_wmma_loss_kernel, dim3(gN, gM), dim3(256), LDS_BYTES, stream,
                     vh, vl, th, tl, log_temp, bias, partials);
  const float scale = (float)(1.0 / ((double)Bv * (double)Bt));
  hipLaunchKernelGGL(finalize_kernel, dim3(1), dim3(256), 0, stream,
                     partials, gM * gN, scale, out);
}
